// TimeToCollisionConstraint_54039278518799
// MI455X (gfx1250) — matrix operand order repaired and checked by reading, compiled, not run on hardware
//
#include <hip/hip_runtime.h>
#include <hip/hip_bf16.h>
#include <math.h>

#define MIN_TTC_F 3.0f
#define BIG_TTC_F 1000000.0f
#define EPS_F 1e-6f

#define Bn 32
#define Nn 256
#define Hn 64
#define NPAIR ((Nn * (Nn - 1)) / 2)   // 32640
#define NTILE32 (NPAIR / 32)          // 1020 (exact)

typedef __bf16 bf16_t;
typedef __attribute__((ext_vector_type(16))) __bf16 v16bf;
typedef __attribute__((ext_vector_type(8)))  __bf16 v8bf;
typedef __attribute__((ext_vector_type(8)))  float  v8f;

// ---------------------------------------------------------------------------
// pair index p -> (i, j) of np.triu_indices(N, k=1), row-major order.
// S(i) = i*(2N-1-i)/2 pairs precede row i.
// ---------------------------------------------------------------------------
__device__ __forceinline__ void pair_from_index(int p, int& i, int& j) {
  const int M = 2 * Nn - 1;  // 511
  float disc = (float)M * (float)M - 8.0f * (float)p;
  int ii = (int)floorf(((float)M - sqrtf(disc)) * 0.5f);
  if (ii < 0) ii = 0;
  if (ii > Nn - 2) ii = Nn - 2;
  while (ii > 0 && (ii * (M - ii)) / 2 > p) --ii;
  while (ii < Nn - 2 && ((ii + 1) * (M - (ii + 1))) / 2 <= p) ++ii;
  i = ii;
  j = ii + 1 + (p - (ii * (M - ii)) / 2);
}

// ---------------------------------------------------------------------------
// Packed-fragment load: each lane's 16 bf16 fragment values are contiguous
// (32 bytes) -> two ds_load_b128 per fragment.
// ---------------------------------------------------------------------------
__device__ __forceinline__ v16bf load_frag_packed(const bf16_t* __restrict__ base,
                                                  int lane) {
  const v8bf* p = (const v8bf*)(base + lane * 16);
  const v8bf lo = p[0];
  const v8bf hi = p[1];
  return __builtin_shufflevector(lo, hi, 0, 1, 2, 3, 4, 5, 6, 7,
                                 8, 9, 10, 11, 12, 13, 14, 15);
}

__device__ __forceinline__ v8f wmma_bf16(v16bf a, v16bf b, v8f c) {
  return __builtin_amdgcn_wmma_f32_16x16x32_bf16(
      /*neg_a=*/false, a, /*neg_b=*/false, b,
      /*c_mod=*/(short)0, c, /*reuse_a=*/false, /*reuse_b=*/false);
}

// ---------------------------------------------------------------------------
// Store one C column (8 rows held by this lane) into the NEXT layer's
// A-fragment-packed buffer, with bias + relu + bf16 convert.
// Value (row m, K=n): kk=n&31; destLane = m + ((kk&16)?16:0);
//                     e = kk&15.
// ---------------------------------------------------------------------------
__device__ __forceinline__ void store_h_packed(bf16_t* __restrict__ hp_kt,
                                               const v8f& c, float bias,
                                               int n, int mb) {
  const int kk = n & 31;
  const int laneAdd = (kk & 16) ? 16 : 0;
  const int e = kk & 15;
  bf16_t* dst = hp_kt + (mb + laneAdd) * 16 + e;
#pragma unroll
  for (int r = 0; r < 8; ++r)
    dst[r * 16] = (bf16_t)fmaxf(c[r] + bias, 0.0f);
}

// ---------------------------------------------------------------------------
// Kernel 0: zero the actions slice of d_out.
// ---------------------------------------------------------------------------
__global__ void zero_actions_kernel(float* __restrict__ out, int n) {
  int idx = blockIdx.x * blockDim.x + threadIdx.x;
  if (idx < n) out[idx] = 0.0f;
}

// ---------------------------------------------------------------------------
// Kernel 1: full (B,N,N) ttc matrix -> d_out ttc slice. 4 j's per thread,
// float4 stores. Streaming 8 MB.
// ---------------------------------------------------------------------------
__global__ void ttc_kernel(const float* __restrict__ pos,
                           const float* __restrict__ vel,
                           float* __restrict__ ttc_out) {
  const unsigned idx = blockIdx.x * blockDim.x + threadIdx.x;
  if (idx >= (unsigned)Bn * Nn * (Nn / 4)) return;
  const int j0 = (idx % (Nn / 4)) * 4;
  const int i  = (idx / (Nn / 4)) % Nn;
  const int b  = idx / ((Nn / 4) * Nn);

  const float* Pi = pos + ((size_t)b * Nn + i) * 3;
  const float* Vi = vel + ((size_t)b * Nn + i) * 3;
  const float pix = Pi[0], piy = Pi[1], piz = Pi[2];
  const float vix = Vi[0], viy = Vi[1], viz = Vi[2];

  float4 out;
  float* o = &out.x;
#pragma unroll
  for (int jj = 0; jj < 4; ++jj) {
    const int j = j0 + jj;
    const float* Pj = pos + ((size_t)b * Nn + j) * 3;
    const float* Vj = vel + ((size_t)b * Nn + j) * 3;
    const float rpx = Pj[0] - pix, rpy = Pj[1] - piy, rpz = Pj[2] - piz;
    const float rvx = Vj[0] - vix, rvy = Vj[1] - viy, rvz = Vj[2] - viz;
    const float dist = sqrtf(rpx * rpx + rpy * rpy + rpz * rpz);
    const float approach = -(rpx * rvx + rpy * rvy + rpz * rvz) / (dist + EPS_F);
    o[jj] = (approach > 0.0f) ? dist / (approach + EPS_F) : BIG_TTC_F;
  }
  *(float4*)(ttc_out + ((size_t)b * Nn + i) * Nn + j0) = out;
}

// ---------------------------------------------------------------------------
// Kernel 2: per-pair activity scale = any_b(ttc < MIN_TTC) ? 0.5 : 0.0.
// Grid over (i,j); closed-form pair index p = S(i) + j - i - 1.
// ---------------------------------------------------------------------------
__global__ void active_kernel(const float* __restrict__ ttc,
                              float* __restrict__ scale) {
  const int idx = blockIdx.x * blockDim.x + threadIdx.x;
  if (idx >= Nn * Nn) return;
  const int j = idx & (Nn - 1);
  const int i = idx >> 8;
  if (j <= i) return;
  const int p = (i * (2 * Nn - 1 - i)) / 2 + (j - i - 1);
  bool any = false;
#pragma unroll 4
  for (int b = 0; b < Bn; ++b)
    any |= (ttc[((size_t)b * Nn + i) * Nn + j] < MIN_TTC_F);
  scale[p] = any ? 0.5f : 0.0f;
}

// ---------------------------------------------------------------------------
// Kernel 3: fused feature-build + 3-layer MLP (bf16 WMMA, f32 accum) +
// scatter-add of +/- avoidance into actions.
// Block = 256 threads = 8 waves. Wave w handles batch (blockIdx.y*8+w) and
// TWO 16-pair row tiles (32 pairs) -> each B fragment feeds 2 WMMAs.
// 28 WMMAs per wave. All LDS traffic is fragment-packed (ds_load_b128).
// ---------------------------------------------------------------------------
__global__ void __launch_bounds__(256)
mlp_kernel(const float* __restrict__ pos, const float* __restrict__ vel,
           const float* __restrict__ W1, const float* __restrict__ b1,
           const float* __restrict__ W2, const float* __restrict__ b2,
           const float* __restrict__ W3, const float* __restrict__ b3,
           const float* __restrict__ scale, float* __restrict__ actions) {
  // B-fragment-packed weights: [kt][nt][lane][e], 16 bf16 contiguous per lane.
  __shared__ __align__(16) bf16_t W1p[4][32][16];      // kt=0 only (K pad 7->32)
  __shared__ __align__(16) bf16_t W2p[2][4][32][16];
  __shared__ __align__(16) bf16_t W3p[2][32][16];      // nt=0 only (N pad 3->16)
  __shared__ float b1s[Hn], b2s[Hn], b3s[16];
  // A-fragment-packed hidden activations per wave: [mt][kt][lane][e].
  __shared__ __align__(16) bf16_t hp[8][2][2][32][16];
  __shared__ int   pis[8][32], pjs[8][32];
  __shared__ float scs[8][32];

  const int t = threadIdx.x;

  __builtin_prefetch(W2, 0, 1);  // global_prefetch_b8: warm weights into GL2

  // --- pack weights into B-fragment order (one 16-elem segment / thread) ---
  // Segment -> (kt, nt, L): lane L covers K = kt*32 + ((L&16)?16:0) + e,
  //                         N = nt*16 + (L&15).
  {
    // W2: 2*4*32 = 256 segments.
    const int L = t & 31, nt = (t >> 5) & 3, kt = t >> 7;
    const int kb = kt * 32 + ((L & 16) ? 16 : 0);
    const int n = nt * 16 + (L & 15);
    bf16_t* dst = &W2p[kt][nt][L][0];
#pragma unroll
    for (int e = 0; e < 16; ++e)
      dst[e] = (bf16_t)W2[(kb + e) * Hn + n];
  }
  if (t < 128) {  // W1: 4*32 = 128 segments (kt = 0).
    const int L = t & 31, nt = t >> 5;
    const int kb = (L & 16) ? 16 : 0;
    const int n = nt * 16 + (L & 15);
    bf16_t* dst = &W1p[nt][L][0];
#pragma unroll
    for (int e = 0; e < 16; ++e) {
      const int k = kb + e;
      dst[e] = (k < 7) ? (bf16_t)W1[k * Hn + n] : (bf16_t)0.0f;
    }
  }
  if (t < 64) {  // W3: 2*32 = 64 segments (nt = 0, N padded to 16).
    const int L = t & 31, kt = t >> 5;
    const int kb = kt * 32 + ((L & 16) ? 16 : 0);
    const int n = L & 15;
    bf16_t* dst = &W3p[kt][L][0];
#pragma unroll
    for (int e = 0; e < 16; ++e)
      dst[e] = (n < 3) ? (bf16_t)W3[(kb + e) * 3 + n] : (bf16_t)0.0f;
  }
  if (t < Hn) { b1s[t] = b1[t]; b2s[t] = b2[t]; }
  if (t < 16) b3s[t] = (t < 3) ? b3[t] : 0.0f;
  __syncthreads();

  const int w    = t >> 5;
  const int lane = t & 31;
  const int b    = blockIdx.y * 8 + w;
  const int nloc = lane & 15;
  const int mb   = (lane & 16) ? 8 : 0;

  // --- feature build: all 32 lanes, one pair each (2 row tiles) ------------
  const int p = blockIdx.x * 32 + lane;
  int pi, pj;
  pair_from_index(p, pi, pj);
  pis[w][lane] = pi;
  pjs[w][lane] = pj;
  scs[w][lane] = scale[p];

  float f[7];
  {
    const float* Pi = pos + ((size_t)b * Nn + pi) * 3;
    const float* Pj = pos + ((size_t)b * Nn + pj) * 3;
    const float* Vi = vel + ((size_t)b * Nn + pi) * 3;
    const float* Vj = vel + ((size_t)b * Nn + pj) * 3;
    f[0] = Pj[0] - Pi[0]; f[1] = Pj[1] - Pi[1]; f[2] = Pj[2] - Pi[2];
    f[3] = Vj[0] - Vi[0]; f[4] = Vj[1] - Vi[1]; f[5] = Vj[2] - Vi[2];
    const float dist = sqrtf(f[0] * f[0] + f[1] * f[1] + f[2] * f[2]);
    const float approach = -(f[0] * f[3] + f[1] * f[4] + f[2] * f[5]) / (dist + EPS_F);
    f[6] = (approach > 0.0f) ? dist / (approach + EPS_F) : BIG_TTC_F;
  }

  // Layer-1 A fragments built IN REGISTERS (ISA 7.12.2):
  // lane L holds row m=L&15, K = kb+e, kb=(L&16)?16:0 (same order as W1p).
  // Features are K=0..6 -> lanes 0..15, e=0..6; lanes 16..31 are zero pad.
  // Tile 0 rows = pairs (lane 0..15 own); tile 1 rows live in lanes 16..31,
  // moved down with __shfl (ds_bpermute).
  v16bf xa0, xa1;
#pragma unroll
  for (int e = 0; e < 16; ++e) { xa0[e] = (bf16_t)0.0f; xa1[e] = (bf16_t)0.0f; }
#pragma unroll
  for (int e = 0; e < 7; ++e) {
    const float fe = __shfl(f[e], 16 + nloc, 32);
    if (lane < 16) {
      xa0[e] = (bf16_t)f[e];
      xa1[e] = (bf16_t)fe;
    }
  }

  // --- Layer 1: h1 = relu(x @ W1 + b1) ------------------------------------
#pragma unroll
  for (int nt = 0; nt < 4; ++nt) {
    const v16bf bf = load_frag_packed(&W1p[nt][0][0], lane);
    v8f c0 = {0.f, 0.f, 0.f, 0.f, 0.f, 0.f, 0.f, 0.f};
    v8f c1 = {0.f, 0.f, 0.f, 0.f, 0.f, 0.f, 0.f, 0.f};
    c0 = wmma_bf16(xa0, bf, c0);
    c1 = wmma_bf16(xa1, bf, c1);
    const int n = nt * 16 + nloc;
    store_h_packed(&hp[w][0][n >> 5][0][0], c0, b1s[n], n, mb);
    store_h_packed(&hp[w][1][n >> 5][0][0], c1, b1s[n], n, mb);
  }

  // --- Layer 2: h2 = relu(h1 @ W2 + b2) -----------------------------------
  {
    const v16bf a00 = load_frag_packed(&hp[w][0][0][0][0], lane);
    const v16bf a01 = load_frag_packed(&hp[w][0][1][0][0], lane);
    const v16bf a10 = load_frag_packed(&hp[w][1][0][0][0], lane);
    const v16bf a11 = load_frag_packed(&hp[w][1][1][0][0], lane);
#pragma unroll
    for (int nt = 0; nt < 4; ++nt) {
      const v16bf bk0 = load_frag_packed(&W2p[0][nt][0][0], lane);
      const v16bf bk1 = load_frag_packed(&W2p[1][nt][0][0], lane);
      v8f c0 = {0.f, 0.f, 0.f, 0.f, 0.f, 0.f, 0.f, 0.f};
      v8f c1 = {0.f, 0.f, 0.f, 0.f, 0.f, 0.f, 0.f, 0.f};
      c0 = wmma_bf16(a00, bk0, c0);
      c0 = wmma_bf16(a01, bk1, c0);
      c1 = wmma_bf16(a10, bk0, c1);
      c1 = wmma_bf16(a11, bk1, c1);
      const int n = nt * 16 + nloc;
      store_h_packed(&hp[w][0][n >> 5][0][0], c0, b2s[n], n, mb);
      store_h_packed(&hp[w][1][n >> 5][0][0], c1, b2s[n], n, mb);
    }
  }

  // --- Layer 3: a = (h2 @ W3 + b3) * scale ; scatter-add into actions ------
  {
    const v16bf a00 = load_frag_packed(&hp[w][0][0][0][0], lane);
    const v16bf a01 = load_frag_packed(&hp[w][0][1][0][0], lane);
    const v16bf a10 = load_frag_packed(&hp[w][1][0][0][0], lane);
    const v16bf a11 = load_frag_packed(&hp[w][1][1][0][0], lane);
    const v16bf bk0 = load_frag_packed(&W3p[0][0][0], lane);
    const v16bf bk1 = load_frag_packed(&W3p[1][0][0], lane);
    v8f c0 = {0.f, 0.f, 0.f, 0.f, 0.f, 0.f, 0.f, 0.f};
    v8f c1 = {0.f, 0.f, 0.f, 0.f, 0.f, 0.f, 0.f, 0.f};
    c0 = wmma_bf16(a00, bk0, c0);
    c0 = wmma_bf16(a01, bk1, c0);
    c1 = wmma_bf16(a10, bk0, c1);
    c1 = wmma_bf16(a11, bk1, c1);

    if (nloc < 3) {
      float* act = actions + (size_t)b * Nn * 3;
      const float bias = b3s[nloc];
#pragma unroll
      for (int mt = 0; mt < 2; ++mt) {
        const v8f& c = mt ? c1 : c0;
#pragma unroll
        for (int r = 0; r < 8; ++r) {
          const int m = mt * 16 + mb + r;
          const float v = (c[r] + bias) * scs[w][m];
          atomicAdd(&act[pis[w][m] * 3 + nloc], -v);
          atomicAdd(&act[pjs[w][m] * 3 + nloc],  v);
        }
      }
    }
  }
}

// ---------------------------------------------------------------------------
extern "C" void kernel_launch(void* const* d_in, const int* in_sizes, int n_in,
                              void* d_out, int out_size, void* d_ws,
                              size_t ws_size, hipStream_t stream) {
  const float* pos = (const float*)d_in[0];
  const float* vel = (const float*)d_in[1];
  const float* W1  = (const float*)d_in[2];
  const float* b1  = (const float*)d_in[3];
  const float* W2  = (const float*)d_in[4];
  const float* b2  = (const float*)d_in[5];
  const float* W3  = (const float*)d_in[6];
  const float* b3  = (const float*)d_in[7];

  float* out_actions = (float*)d_out;                       // (B, N, 3)
  float* out_ttc     = out_actions + (size_t)Bn * Nn * 3;   // (B, N, N)
  float* scale       = (float*)d_ws;                        // (NPAIR,)

  const int nAct = Bn * Nn * 3;
  zero_actions_kernel<<<(nAct + 255) / 256, 256, 0, stream>>>(out_actions, nAct);

  const unsigned nTtc4 = (unsigned)Bn * Nn * (Nn / 4);
  ttc_kernel<<<(nTtc4 + 255) / 256, 256, 0, stream>>>(pos, vel, out_ttc);

  active_kernel<<<(Nn * Nn + 255) / 256, 256, 0, stream>>>(out_ttc, scale);

  dim3 grid(NTILE32, Bn / 8, 1);
  mlp_kernel<<<grid, 256, 0, stream>>>(pos, vel, W1, b1, W2, b2, W3, b3,
                                       scale, out_actions);
}
